// AttLstm_90640989815589
// MI455X (gfx1250) — matrix operand order repaired and checked by reading, not compiled, not run on hardware
//
#include <hip/hip_runtime.h>

// Problem sizes (compile-time constants from the reference)
#define BB 512   // batch
#define TT 256   // timesteps
#define II 256   // input features
#define HH 512   // hidden
#define GG 2048  // 4*H gates

typedef __attribute__((ext_vector_type(16))) __bf16 v16bf;
typedef __attribute__((ext_vector_type(8)))  __bf16 v8bf;
typedef __attribute__((ext_vector_type(8)))  float  v8f;
typedef int v4i __attribute__((vector_size(16)));

#if __has_builtin(__builtin_amdgcn_global_load_async_to_lds_b128)
#define ASYNC_STAGE 1
// generic ptr -> addrspace-qualified int4-vector ptr (two-step: AS cast, then type cast)
#define AS_GLOBAL_V4I(p) \
  ((__attribute__((address_space(1))) v4i*)(__attribute__((address_space(1))) void*)(void*)(p))
#define AS_LDS_V4I(p) \
  ((__attribute__((address_space(3))) v4i*)(__attribute__((address_space(3))) void*)(void*)(p))
#else
#define ASYNC_STAGE 0
#endif

__device__ __forceinline__ __bf16 f2bf(float f) {
  unsigned u = __builtin_bit_cast(unsigned, f);
  unsigned r = u + 0x7FFFu + ((u >> 16) & 1u);   // round-to-nearest-even
  unsigned short s = (unsigned short)(r >> 16);
  return __builtin_bit_cast(__bf16, s);
}
__device__ __forceinline__ float bf2f(__bf16 b) {
  unsigned short s = __builtin_bit_cast(unsigned short, b);
  unsigned u = ((unsigned)s) << 16;
  return __builtin_bit_cast(float, u);
}
__device__ __forceinline__ float sigm(float x) { return 1.0f / (1.0f + __expf(-x)); }
__device__ __forceinline__ float tanh_(float x) { return 1.0f - 2.0f / (__expf(2.0f * x) + 1.0f); }

// Load a 16-bit A fragment row slice (two 16B chunks per ISA A layout).
__device__ __forceinline__ v16bf load_a(const __bf16* __restrict__ row, int k0) {
  v16bf a;
  *((v8bf*)&a)       = *(const v8bf*)(row + k0);
  *(((v8bf*)&a) + 1) = *(const v8bf*)(row + k0 + 16);
  return a;
}

// Stage this block's B fragments (NKT k-tiles x 4 gate tiles, 1KB each) into LDS.
// Uses gfx1250 async global->LDS copies when available.
template <int NKT>
__device__ __forceinline__ void stage_b(const __bf16* __restrict__ wfrag, int ht,
                                        __bf16* sm) {
  const int tid = threadIdx.x;
#pragma unroll 4
  for (int u = tid; u < NKT * 4 * 64; u += 256) {   // 16B units
    const int chunk = u >> 6;                       // kt*4 + g
    const int off   = (u & 63) * 8;                 // element offset within 1KB chunk
    const int kt = chunk >> 2, g = chunk & 3;
    const __bf16* src = wfrag + (((size_t)(kt * 128 + g * 32 + ht)) << 9) + off;
    __bf16* dst = sm + chunk * 512 + off;
#if ASYNC_STAGE
    __builtin_amdgcn_global_load_async_to_lds_b128(AS_GLOBAL_V4I(src), AS_LDS_V4I(dst), 0, 0);
#else
    *(uint4*)dst = *(const uint4*)src;
#endif
  }
#if ASYNC_STAGE
#if __has_builtin(__builtin_amdgcn_s_wait_asynccnt)
  __builtin_amdgcn_s_wait_asynccnt(0);
#else
  asm volatile("s_wait_asynccnt 0x0" ::: "memory");
#endif
#endif
  __syncthreads();
}

// ---------------- prep kernels ----------------

__global__ void k_cvt_x(const float* __restrict__ x, __bf16* __restrict__ xb, int n) {
  int i = blockIdx.x * blockDim.x + threadIdx.x;
  if (i < n) xb[i] = f2bf(x[i]);
}

// Pack W (GG x K, row-major f32) into WMMA B-fragment tiles.
// Tile (kt,nt): B[k][n] = W[n][k]. Per-lane (ISA 16-bit B, wave32):
// lane L -> N = L%16, K-half = L/16; element e -> K = 8*(L/16)+(e%8)+16*(e/8).
__global__ void k_pack_w(const float* __restrict__ W, __bf16* __restrict__ frag,
                         int K, int total) {
  int o = blockIdx.x * blockDim.x + threadIdx.x;
  if (o >= total) return;
  int w    = o & 511;
  int tile = o >> 9;
  int nt   = tile & 127;     // GG/16 = 128 n-tiles
  int kt   = tile >> 7;
  int lane = w >> 4;
  int e    = w & 15;
  int n = nt * 16 + (lane & 15);
  int k = kt * 32 + (lane >> 4) * 8 + (e & 7) + (e >> 3) * 16;
  frag[o] = f2bf(W[(size_t)n * K + k]);
}

__global__ void k_init(const float* __restrict__ bih, const float* __restrict__ bhh,
                       float* __restrict__ bias, __bf16* __restrict__ h0,
                       float* __restrict__ c0) {
  int i = blockIdx.x * blockDim.x + threadIdx.x;
  if (i < GG) bias[i] = bih[i] + bhh[i];
  if (i < BB * HH) { h0[i] = f2bf(0.0f); c0[i] = 0.0f; }
}

// ---------------- precompute: XW = x @ W_ih^T + b (all timesteps, parallel) ----
// Block: fixed ht, 8 waves cover 8 row-tiles; W_ih fragments staged in LDS (32KB).
__global__ __launch_bounds__(256) void k_xw(const __bf16* __restrict__ xb,
                                            const __bf16* __restrict__ wih,
                                            const float* __restrict__ bias,
                                            __bf16* __restrict__ xw) {
  __shared__ alignas(32) __bf16 bsm[(II / 32) * 4 * 512];  // 32 KB
  const int ht = blockIdx.x & 31;                          // block-constant
  const int mt = (blockIdx.x >> 5) * 8 + (threadIdx.x >> 5);
  const int lane  = threadIdx.x & 31;
  const int laneM = lane & 15;
  const int laneH = lane >> 4;

  stage_b<II / 32>(wih, ht, bsm);

  v8f acc[4];
#pragma unroll
  for (int g = 0; g < 4; ++g) acc[g] = {};

  const int r = mt * 16 + laneM;
  const __bf16* arow = xb + (size_t)r * II;

  v16bf aC = load_a(arow, laneH * 8);
#pragma unroll
  for (int kt = 0; kt < II / 32; ++kt) {
    v16bf aN;
    if (kt < II / 32 - 1) aN = load_a(arow, (kt + 1) * 32 + laneH * 8);
#pragma unroll
    for (int g = 0; g < 4; ++g) {
      const v16bf b = *(const v16bf*)(bsm + (kt * 4 + g) * 512 + lane * 16);
      acc[g] = __builtin_amdgcn_wmma_f32_16x16x32_bf16(false, aC, false, b,
                                                       (short)0, acc[g], false, false);
    }
    if (kt < II / 32 - 1) aC = aN;
  }

  const int hcol = ht * 16 + laneM;
#pragma unroll
  for (int g = 0; g < 4; ++g) {
    const float bv = bias[g * 512 + hcol];
#pragma unroll
    for (int v = 0; v < 8; ++v) {
      const int rr = mt * 16 + v + laneH * 8;
      xw[(size_t)rr * GG + g * 512 + hcol] = f2bf(acc[g][v] + bv);
    }
  }
}

// ---------------- LSTM step with precomputed input projection ----------------
// Only h @ W_hh^T (16 k-tiles) sequential; W_hh fragments staged in LDS (64KB).
__global__ __launch_bounds__(256) void k_step_pre(
    const __bf16* __restrict__ whh,  const __bf16* __restrict__ xw,
    const __bf16* __restrict__ hin,  float* __restrict__ cst,
    __bf16* __restrict__ hout,       __bf16* __restrict__ hstates, int t) {
  __shared__ alignas(32) __bf16 bsm[(HH / 32) * 4 * 512];  // 64 KB
  const int ht = blockIdx.x & 31;
  const int mt = (blockIdx.x >> 5) * 8 + (threadIdx.x >> 5);
  const int lane  = threadIdx.x & 31;
  const int laneM = lane & 15;
  const int laneH = lane >> 4;

  stage_b<HH / 32>(whh, ht, bsm);

  v8f acc[4];
#pragma unroll
  for (int g = 0; g < 4; ++g) acc[g] = {};

  const int bIdx = mt * 16 + laneM;
  const __bf16* hrow = hin + (size_t)bIdx * HH;

  v16bf aC = load_a(hrow, laneH * 8);
#pragma unroll
  for (int kt = 0; kt < HH / 32; ++kt) {
    v16bf aN;
    if (kt < HH / 32 - 1) aN = load_a(hrow, (kt + 1) * 32 + laneH * 8);
#pragma unroll
    for (int g = 0; g < 4; ++g) {
      const v16bf b = *(const v16bf*)(bsm + (kt * 4 + g) * 512 + lane * 16);
      acc[g] = __builtin_amdgcn_wmma_f32_16x16x32_bf16(false, aC, false, b,
                                                       (short)0, acc[g], false, false);
    }
    if (kt < HH / 32 - 1) aC = aN;
  }

  const int hcol = ht * 16 + laneM;
#pragma unroll
  for (int v = 0; v < 8; ++v) {
    const int brow = mt * 16 + v + laneH * 8;
    const __bf16* xwr = xw + (size_t)(brow * TT + t) * GG + hcol;  // bias folded in
    const float iv = sigm(acc[0][v] + bf2f(xwr[0]));
    const float fv = sigm(acc[1][v] + bf2f(xwr[512]));
    const float gv = tanh_(acc[2][v] + bf2f(xwr[1024]));
    const float ov = sigm(acc[3][v] + bf2f(xwr[1536]));
    const size_t ci = (size_t)brow * HH + hcol;
    const float cn = fv * cst[ci] + iv * gv;
    cst[ci] = cn;
    const float hn = ov * tanh_(cn);
    const __bf16 hb = f2bf(hn);
    hout[ci] = hb;
    hstates[((size_t)brow * TT + t) * HH + hcol] = hb;
  }
}

// ---------------- fused LSTM step (fallback when ws too small for xw) --------
// h-phase B staged in LDS; x-phase B streamed from global (register pipelined).
__global__ __launch_bounds__(256) void k_step(
    const __bf16* __restrict__ xb,   const __bf16* __restrict__ wih,
    const __bf16* __restrict__ whh,  const float* __restrict__ bias,
    const __bf16* __restrict__ hin,  float* __restrict__ cst,
    __bf16* __restrict__ hout,       __bf16* __restrict__ hstates, int t) {
  __shared__ alignas(32) __bf16 bsm[(HH / 32) * 4 * 512];  // 64 KB
  const int ht = blockIdx.x & 31;
  const int mt = (blockIdx.x >> 5) * 8 + (threadIdx.x >> 5);
  const int lane  = threadIdx.x & 31;
  const int laneM = lane & 15;
  const int laneH = lane >> 4;

  stage_b<HH / 32>(whh, ht, bsm);

  v8f acc[4];
#pragma unroll
  for (int g = 0; g < 4; ++g) acc[g] = {};

  const int bIdx = mt * 16 + laneM;
  const __bf16* xrow = xb + ((size_t)bIdx * TT + t) * II;
  const __bf16* hrow = hin + (size_t)bIdx * HH;

  // x phase: B from global, register double-buffered
  {
    auto fetch = [&](int kt, v16bf& a, v16bf* b) {
      a = load_a(xrow, kt * 32 + laneH * 8);
#pragma unroll
      for (int g = 0; g < 4; ++g)
        b[g] = *(const v16bf*)(wih + (((size_t)(kt * 128 + g * 32 + ht)) << 9) + lane * 16);
    };
    v16bf aC, bC[4];
    fetch(0, aC, bC);
#pragma unroll
    for (int kt = 0; kt < II / 32; ++kt) {
      v16bf aN, bN[4];
      if (kt < II / 32 - 1) fetch(kt + 1, aN, bN);
#pragma unroll
      for (int g = 0; g < 4; ++g)
        acc[g] = __builtin_amdgcn_wmma_f32_16x16x32_bf16(false, aC, false, bC[g],
                                                         (short)0, acc[g], false, false);
      if (kt < II / 32 - 1) {
        aC = aN;
#pragma unroll
        for (int g = 0; g < 4; ++g) bC[g] = bN[g];
      }
    }
  }

  // h phase: B from LDS
  v16bf aC = load_a(hrow, laneH * 8);
#pragma unroll
  for (int kt = 0; kt < HH / 32; ++kt) {
    v16bf aN;
    if (kt < HH / 32 - 1) aN = load_a(hrow, (kt + 1) * 32 + laneH * 8);
#pragma unroll
    for (int g = 0; g < 4; ++g) {
      const v16bf b = *(const v16bf*)(bsm + (kt * 4 + g) * 512 + lane * 16);
      acc[g] = __builtin_amdgcn_wmma_f32_16x16x32_bf16(false, aC, false, b,
                                                       (short)0, acc[g], false, false);
    }
    if (kt < HH / 32 - 1) aC = aN;
  }

  const int hcol = ht * 16 + laneM;
  const float bi = bias[hcol];
  const float bf = bias[hcol + 512];
  const float bg = bias[hcol + 1024];
  const float bo = bias[hcol + 1536];
#pragma unroll
  for (int v = 0; v < 8; ++v) {
    const int brow = mt * 16 + v + laneH * 8;
    const float iv = sigm(acc[0][v] + bi);
    const float fv = sigm(acc[1][v] + bf);
    const float gv = tanh_(acc[2][v] + bg);
    const float ov = sigm(acc[3][v] + bo);
    const size_t ci = (size_t)brow * HH + hcol;
    const float cn = fv * cst[ci] + iv * gv;
    cst[ci] = cn;
    const float hn = ov * tanh_(cn);
    const __bf16 hb = f2bf(hn);
    hout[ci] = hb;
    hstates[((size_t)brow * TT + t) * HH + hcol] = hb;
  }
}

// ---------------- attention pooling ----------------

__global__ __launch_bounds__(256) void k_score(const __bf16* __restrict__ hs,
                                               float* __restrict__ score) {
  const int gw   = blockIdx.x * 8 + (threadIdx.x >> 5);  // one wave per (b,t)
  const int lane = threadIdx.x & 31;
  const int b = gw / TT, t = gw % TT;
  const __bf16* hrow  = hs + ((size_t)b * TT + t) * HH;
  const __bf16* hlast = hs + ((size_t)b * TT + (TT - 1)) * HH;
  float s = 0.0f;
#pragma unroll
  for (int it = 0; it < HH / (32 * 8); ++it) {
    const v8bf a  = *(const v8bf*)(hrow  + it * 256 + lane * 8);
    const v8bf hl = *(const v8bf*)(hlast + it * 256 + lane * 8);
#pragma unroll
    for (int j = 0; j < 8; ++j) s += bf2f(a[j]) * bf2f(hl[j]);
  }
#pragma unroll
  for (int m = 16; m; m >>= 1) s += __shfl_xor(s, m, 32);
  if (lane == 0) score[gw] = s;
}

__global__ __launch_bounds__(256) void k_sum(const float* __restrict__ score,
                                             float* __restrict__ ssum) {
  const int gw   = blockIdx.x * 8 + (threadIdx.x >> 5);  // one wave per b
  const int lane = threadIdx.x & 31;
  float s = 0.0f;
  for (int t = lane; t < TT; t += 32) s += score[gw * TT + t];
#pragma unroll
  for (int m = 16; m; m >>= 1) s += __shfl_xor(s, m, 32);
  if (lane == 0) ssum[gw] = s;
}

__global__ __launch_bounds__(256) void k_ctx(const __bf16* __restrict__ hs,
                                             const float* __restrict__ score,
                                             const float* __restrict__ ssum,
                                             float* __restrict__ out) {
  const int i = blockIdx.x * blockDim.x + threadIdx.x;  // < BB*HH, coalesced in h
  const int b = i >> 9;
  const int h = i & (HH - 1);
  const float inv = 1.0f / ssum[b];
  const float* sc = score + b * TT;
  const __bf16* hp = hs + (size_t)b * TT * HH + h;
  float acc = 0.0f;
  for (int t = 0; t < TT; ++t) acc += sc[t] * bf2f(hp[(size_t)t * HH]);
  out[i] = acc * inv;
}

// ---------------- launcher ----------------

extern "C" void kernel_launch(void* const* d_in, const int* in_sizes, int n_in,
                              void* d_out, int out_size, void* d_ws, size_t ws_size,
                              hipStream_t stream) {
  const float* x    = (const float*)d_in[0];
  const float* Wih  = (const float*)d_in[1];
  const float* Whh  = (const float*)d_in[2];
  const float* bih  = (const float*)d_in[3];
  const float* bhh  = (const float*)d_in[4];
  float* out = (float*)d_out;

  char* ws = (char*)d_ws;
  // workspace layout (all offsets 256B-aligned)
  __bf16* xb   = (__bf16*)(ws + 0);                         // 64 MB
  __bf16* wihf = (__bf16*)(ws + (size_t)67108864);          // 1 MB
  __bf16* whhf = (__bf16*)(ws + (size_t)68157440);          // 2 MB
  float*  bias = (float*) (ws + (size_t)70254592);          // 8 KB
  __bf16* hA   = (__bf16*)(ws + (size_t)70262784);          // 512 KB
  __bf16* hB   = (__bf16*)(ws + (size_t)70787072);          // 512 KB
  float*  cst  = (float*) (ws + (size_t)71311360);          // 1 MB
  __bf16* hst  = (__bf16*)(ws + (size_t)72359936);          // 128 MB
  float*  scr  = (float*) (ws + (size_t)206577664);         // 512 KB
  float*  ssum = (float*) (ws + (size_t)207101952);         // 2 KB
  const size_t XW_OFF  = 207104000;                         // xw: 512 MB (optional)
  const size_t XW_SIZE = (size_t)BB * TT * GG * 2;
  __bf16* xw = (__bf16*)(ws + XW_OFF);
  const bool use_pre = (ws_size >= XW_OFF + XW_SIZE);       // constant -> deterministic

  // prep: convert x, pack weight fragments, bias + zero state
  {
    const int n = BB * TT * II;
    k_cvt_x<<<(n + 255) / 256, 256, 0, stream>>>(x, xb, n);
  }
  k_pack_w<<<(II / 32) * 128 * 512 / 256, 256, 0, stream>>>(Wih, wihf, II, (II / 32) * 128 * 512);
  k_pack_w<<<(HH / 32) * 128 * 512 / 256, 256, 0, stream>>>(Whh, whhf, HH, (HH / 32) * 128 * 512);
  k_init<<<(BB * HH + 255) / 256, 256, 0, stream>>>(bih, bhh, bias, hA, cst);

  if (use_pre) {
    // all input projections as one big parallel GEMM (off the critical path)
    k_xw<<<32 * (BB * TT / 16 / 8), 256, 0, stream>>>(xb, wihf, bias, xw);
    for (int t = 0; t < TT; ++t) {
      const __bf16* hin = (t & 1) ? hB : hA;
      __bf16* hout      = (t & 1) ? hA : hB;
      k_step_pre<<<128, 256, 0, stream>>>(whhf, xw, hin, cst, hout, hst, t);
    }
  } else {
    for (int t = 0; t < TT; ++t) {
      const __bf16* hin = (t & 1) ? hB : hA;
      __bf16* hout      = (t & 1) ? hA : hB;
      k_step<<<128, 256, 0, stream>>>(xb, wihf, whhf, bias, hin, cst, hout, hst, t);
    }
  }

  // attention pooling
  k_score<<<BB * TT / 8, 256, 0, stream>>>(hst, scr);
  k_sum<<<BB / 8, 256, 0, stream>>>(scr, ssum);
  k_ctx<<<BB * HH / 256, 256, 0, stream>>>(hst, scr, ssum, out);
}
